// EdgeMLPMPN_9440338117362
// MI455X (gfx1250) — matrix operand order repaired and checked by reading, not compiled, not run on hardware
//
#include <hip/hip_runtime.h>
#include <hip/hip_bf16.h>

// ---------------------------------------------------------------------------
// EdgeMLP on MI455X (gfx1250): bf16 WMMA pipeline, fragment-major LDS.
//   59 GFLOP, x-table (51MB) resident in the 192MB L2 -> matrix-core bound.
//   1 wave computes 16 edges; weights live in LDS pre-swizzled so every
//   WMMA operand fragment is a contiguous 32B per-lane ds_load_b128 pair.
// ---------------------------------------------------------------------------

typedef __attribute__((ext_vector_type(16))) __bf16 v16bf;
typedef __attribute__((ext_vector_type(8)))  float  v8f;

#define IN_CH 128
#define HID   128
#define K1    256              // 2*IN_CH (concat of src/tgt features)
#define EDGES_PER_WAVE  16
#define WAVES_PER_BLOCK 8
#define EDGES_PER_BLOCK 128
#define NFRAG1 64              // (K1/32) * (HID/16)
#define NFRAG2 32              // (HID/32) * (HID/16)
#define MAX_BLOCKS 1024

struct SMem {
    __bf16 W1s[NFRAG1 * 32 * 16];                 // 64 KB fragment-major
    __bf16 W2s[NFRAG2 * 32 * 16];                 // 32 KB fragment-major
    float  B1[HID];
    float  B2[HID];
    float  W3[HID];
    __bf16 H1[WAVES_PER_BLOCK][4 * 32 * 16];      // 32 KB, A-frag layout/wave
};

__device__ __forceinline__ float elu1(float v) {
    return v > 0.0f ? v : (__expf(v) - 1.0f);
}

// Swizzle a row-major [K][128] f32 weight matrix into fragment-major bf16:
//   frag = kt*8 + nt; per (frag, lane): 16 contiguous bf16 =
//   W[kt*32 + (lane&16?8:0) + (i<8?i:i+8)][nt*16 + (lane&15)], i = 0..15
// (matches the documented 16-bit B VGPR layout for 16x16x32 WMMA).
__device__ __forceinline__ void stage_wfrag(__bf16* dst, const float* gW,
                                            int nFragLanes, int tid) {
    for (int fl = tid; fl < nFragLanes; fl += 256) {
        const int lane = fl & 31;
        const int frag = fl >> 5;
        const int kt   = frag >> 3;
        const int nt   = frag & 7;
        const int krow = kt * 32 + ((lane & 16) ? 8 : 0);
        const int c    = nt * 16 + (lane & 15);
        const float* g = gW + krow * HID + c;
        v16bf t;
#pragma unroll
        for (int i = 0; i < 16; ++i) t[i] = (__bf16)g[(i < 8 ? i : i + 8) * HID];
        *(v16bf*)(dst + fl * 16) = t;             // one 32B LDS store
    }
}

// A-fragment packing from gathered f32: elements 0..7 = K-run (off..off+7),
// elements 8..15 = K-run (off+16..off+23), off = hi?8:0.
__device__ __forceinline__ v16bf pack_a(float4 r0, float4 r1, float4 r2, float4 r3) {
    v16bf a;
    a[0]=(__bf16)r0.x;  a[1]=(__bf16)r0.y;  a[2]=(__bf16)r0.z;  a[3]=(__bf16)r0.w;
    a[4]=(__bf16)r1.x;  a[5]=(__bf16)r1.y;  a[6]=(__bf16)r1.z;  a[7]=(__bf16)r1.w;
    a[8]=(__bf16)r2.x;  a[9]=(__bf16)r2.y;  a[10]=(__bf16)r2.z; a[11]=(__bf16)r2.w;
    a[12]=(__bf16)r3.x; a[13]=(__bf16)r3.y; a[14]=(__bf16)r3.z; a[15]=(__bf16)r3.w;
    return a;
}

__global__ __launch_bounds__(256, 1)
void edge_mlp_wmma(const float* __restrict__ x,
                   const long long* __restrict__ ei,   // [2, nEdges] int64
                   const float* __restrict__ gW1, const float* __restrict__ gb1,
                   const float* __restrict__ gW2, const float* __restrict__ gb2,
                   const float* __restrict__ gW3, const float* __restrict__ gb3,
                   float* __restrict__ out, int nEdges)
{
    __shared__ SMem sm;
    const int tid = threadIdx.x;

    // ---- once per (persistent) block: swizzle weights into LDS ----
    stage_wfrag(sm.W1s, gW1, NFRAG1 * 32, tid);
    stage_wfrag(sm.W2s, gW2, NFRAG2 * 32, tid);
    if (tid < HID) {
        sm.B1[tid] = gb1[tid];
        sm.B2[tid] = gb2[tid];
        sm.W3[tid] = gW3[tid];
    }
    __syncthreads();

    const int wave = tid >> 5;
    const int lane = tid & 31;
    const int col  = lane & 15;
    const bool hi  = lane >= 16;
    const int hioff8 = hi ? 8 : 0;
    const int colLo  = col & 7;                   // h1 writeback addressing
    const int colHi16 = (col & 8) ? 16 : 0;

    const v16bf* W1f = (const v16bf*)sm.W1s;
    const v16bf* W2f = (const v16bf*)sm.W2s;
    __bf16*      h1  = sm.H1[wave];
    const v16bf* h1f = (const v16bf*)h1;
    const float  b3v = gb3[0];

    const int nChunks = (nEdges + EDGES_PER_BLOCK - 1) / EDGES_PER_BLOCK;
    for (int chunk = blockIdx.x; chunk < nChunks; chunk += gridDim.x) {
        long long e0 = (long long)chunk * EDGES_PER_BLOCK
                     + (long long)wave * EDGES_PER_WAVE;
        // Clamp tail (duplicate identical work) so EXEC stays all-ones.
        if (e0 + EDGES_PER_WAVE > nEdges) e0 = (long long)nEdges - EDGES_PER_WAVE;

        const long long srcN = ei[e0 + col];
        const long long tgtN = ei[(long long)nEdges + e0 + col];
        const float* rowS = x + srcN * IN_CH;
        const float* rowT = x + tgtN * IN_CH;
        __builtin_prefetch((const void*)rowT, 0, 3);   // global_prefetch_b8

        // ============ layer 1: [16x256] x [256x128] + b1 ============
        v8f acc[8];
#pragma unroll
        for (int nt = 0; nt < 8; ++nt) {
            const float bv = sm.B1[nt * 16 + col];
#pragma unroll
            for (int r = 0; r < 8; ++r) acc[nt][r] = bv;
        }
#pragma unroll
        for (int kt = 0; kt < 8; ++kt) {
            const int k = kt * 32;
            const float* rp = (k < IN_CH) ? (rowS + k) : (rowT + (k - IN_CH));
            const int off = hi ? 8 : 0;
            float4 r0 = *(const float4*)(rp + off + 0);
            float4 r1 = *(const float4*)(rp + off + 4);
            float4 r2 = *(const float4*)(rp + off + 16);
            float4 r3 = *(const float4*)(rp + off + 20);
            v16bf a = pack_a(r0, r1, r2, r3);
#pragma unroll
            for (int nt = 0; nt < 8; ++nt) {
                v16bf b = W1f[(kt * 8 + nt) * 32 + lane];   // 2x ds_load_b128
                acc[nt] = __builtin_amdgcn_wmma_f32_16x16x32_bf16(
                    false, a, false, b, (short)0, acc[nt], false, false);
            }
        }

        // ---- ELU + restage h1 directly in A-fragment layout ----
        // value (M = r+hioff8, c = nt*16+col) ->
        //   [kt2 = nt>>1][readerLane = M + colHi16][elem = colLo + 8*(nt&1)]
#pragma unroll
        for (int nt = 0; nt < 8; ++nt) {
#pragma unroll
            for (int r = 0; r < 8; ++r) {
                float v = elu1(acc[nt][r]);
                const int off = (nt >> 1) * 512
                              + (r + hioff8 + colHi16) * 16
                              + colLo + ((nt & 1) << 3);
                h1[off] = (__bf16)v;
            }
        }
        // In-wave cross-lane handoff through LDS: drain DS stores first.
        asm volatile("s_wait_dscnt 0" ::: "memory");

        // ============ layer 2: [16x128] x [128x128] + b2 ============
        v8f acc2[8];
#pragma unroll
        for (int nt = 0; nt < 8; ++nt) {
            const float bv = sm.B2[nt * 16 + col];
#pragma unroll
            for (int r = 0; r < 8; ++r) acc2[nt][r] = bv;
        }
#pragma unroll
        for (int kt = 0; kt < 4; ++kt) {
            v16bf a = h1f[kt * 32 + lane];                  // 2x ds_load_b128
#pragma unroll
            for (int nt = 0; nt < 8; ++nt) {
                v16bf b = W2f[(kt * 8 + nt) * 32 + lane];   // 2x ds_load_b128
                acc2[nt] = __builtin_amdgcn_wmma_f32_16x16x32_bf16(
                    false, a, false, b, (short)0, acc2[nt], false, false);
            }
        }

        // ============ layer 3: ELU then dot with W3 (N=1) on VALU ============
        float p[8];
#pragma unroll
        for (int r = 0; r < 8; ++r) p[r] = 0.0f;
#pragma unroll
        for (int nt = 0; nt < 8; ++nt) {
            const float w3 = sm.W3[nt * 16 + col];
#pragma unroll
            for (int r = 0; r < 8; ++r) p[r] += elu1(acc2[nt][r]) * w3;
        }
        // Reduce over the 16 column-lanes of each half (masks stay in-group).
#pragma unroll
        for (int m = 1; m < 16; m <<= 1) {
#pragma unroll
            for (int r = 0; r < 8; ++r) p[r] += __shfl_xor(p[r], m, 32);
        }
        if (col == 0) {
#pragma unroll
            for (int r = 0; r < 8; ++r) out[e0 + hioff8 + r] = p[r] + b3v;
        }
    }
}

extern "C" void kernel_launch(void* const* d_in, const int* in_sizes, int n_in,
                              void* d_out, int out_size, void* d_ws, size_t ws_size,
                              hipStream_t stream) {
    const float*     x   = (const float*)d_in[0];
    const long long* ei  = (const long long*)d_in[1];
    const float*     W1  = (const float*)d_in[2];
    const float*     b1  = (const float*)d_in[3];
    const float*     W2  = (const float*)d_in[4];
    const float*     b2  = (const float*)d_in[5];
    const float*     W3  = (const float*)d_in[6];
    const float*     b3  = (const float*)d_in[7];
    float* out = (float*)d_out;

    const int nEdges  = in_sizes[1] / 2;
    const int nChunks = (nEdges + EDGES_PER_BLOCK - 1) / EDGES_PER_BLOCK;
    const int blocks  = nChunks < MAX_BLOCKS ? nChunks : MAX_BLOCKS;
    hipLaunchKernelGGL(edge_mlp_wmma, dim3(blocks), dim3(256), 0, stream,
                       x, ei, W1, b1, W2, b2, W3, b3, out, nEdges);
}
